// RotaryMultiheadAttention_85495618995178
// MI455X (gfx1250) — matrix operand order repaired and checked by reading, not compiled, not run on hardware
//
#include <hip/hip_runtime.h>

#define B_   4
#define T_   2048
#define E_   1024
#define H_   16
#define D_   64
#define F_   32
#define BT_  (B_*T_)              // 8192
#define BHTD (B_*H_*T_*D_)        // 8388608

typedef __attribute__((ext_vector_type(16))) __bf16 v16bf;
typedef __attribute__((ext_vector_type(8)))  float  v8f;
typedef int v4i_vec __attribute__((vector_size(16)));   // matches builtin param

union FragBF { v16bf v; uint4 q[2]; };

__device__ __forceinline__ unsigned short f2bf(float f) {
  unsigned u = __float_as_uint(f);
  unsigned r = 0x7FFFu + ((u >> 16) & 1u);   // round-to-nearest-even
  return (unsigned short)((u + r) >> 16);
}
__device__ __forceinline__ float bf2f(unsigned short h) {
  return __uint_as_float(((unsigned)h) << 16);
}

// ---------------------------------------------------------------------------
// Async global->LDS staging (CDNA5 GLOBAL_LOAD_ASYNC_TO_LDS_B128, ASYNCcnt).
// No VGPR round-trip; completion enforced with s_wait_asynccnt before the
// workgroup barrier. Falls back to sync load+ds_store if the builtin is
// absent on this toolchain.
// ---------------------------------------------------------------------------
#if __has_builtin(__builtin_amdgcn_global_load_async_to_lds_b128) && \
    __has_builtin(__builtin_amdgcn_s_wait_asynccnt)
#define HAVE_ASYNC_LDS 1
#else
#define HAVE_ASYNC_LDS 0
#endif

__device__ __forceinline__ void stage16_async(const unsigned short* g,
                                              unsigned short* l) {
#if HAVE_ASYNC_LDS
  // generic LDS pointer -> 32-bit LDS offset (ISA: flat addr low 32 bits)
  __builtin_amdgcn_global_load_async_to_lds_b128(
      (__attribute__((address_space(1))) v4i_vec*)(unsigned long long)(uintptr_t)g,
      (__attribute__((address_space(3))) v4i_vec*)(unsigned)(uintptr_t)l,
      0, 0);
#else
  *(uint4*)l = *(const uint4*)g;
#endif
}
__device__ __forceinline__ void stage_wait() {
#if HAVE_ASYNC_LDS
  __builtin_amdgcn_s_wait_asynccnt(0);
#endif
}

// DPP16 lane permute within a 16-lane row (wave32: two independent rows).
// quad_perm xor1/xor2 + row_ror:4/8 form a rotate-butterfly that fully
// reduces each 16-lane half without touching LDS.
template <int CTRL>
__device__ __forceinline__ float dppmov(float x) {
  return __int_as_float(__builtin_amdgcn_update_dpp(
      0, __float_as_int(x), CTRL, 0xF, 0xF, true));
}
#define DPP_XOR1 0xB1   // quad_perm [1,0,3,2]
#define DPP_XOR2 0x4E   // quad_perm [2,3,0,1]
#define DPP_ROR4 0x124  // row_ror:4
#define DPP_ROR8 0x128  // row_ror:8

__device__ __forceinline__ void red16_max(float (&v)[8]) {
#pragma unroll
  for (int r = 0; r < 8; ++r) v[r] = fmaxf(v[r], dppmov<DPP_XOR1>(v[r]));
#pragma unroll
  for (int r = 0; r < 8; ++r) v[r] = fmaxf(v[r], dppmov<DPP_XOR2>(v[r]));
#pragma unroll
  for (int r = 0; r < 8; ++r) v[r] = fmaxf(v[r], dppmov<DPP_ROR4>(v[r]));
#pragma unroll
  for (int r = 0; r < 8; ++r) v[r] = fmaxf(v[r], dppmov<DPP_ROR8>(v[r]));
}
__device__ __forceinline__ void red16_sum(float (&v)[8]) {
#pragma unroll
  for (int r = 0; r < 8; ++r) v[r] += dppmov<DPP_XOR1>(v[r]);
#pragma unroll
  for (int r = 0; r < 8; ++r) v[r] += dppmov<DPP_XOR2>(v[r]);
#pragma unroll
  for (int r = 0; r < 8; ++r) v[r] += dppmov<DPP_ROR4>(v[r]);
#pragma unroll
  for (int r = 0; r < 8; ++r) v[r] += dppmov<DPP_ROR8>(v[r]);
}

// ---------------------------------------------------------------------------
// fp32 -> bf16 elementwise (vectorized by 4)
// ---------------------------------------------------------------------------
__global__ __launch_bounds__(256) void f32_to_bf16_kernel(
    const float* __restrict__ in, unsigned short* __restrict__ out, int n4) {
  int i = blockIdx.x * blockDim.x + threadIdx.x;
  if (i < n4) {
    float4 v = ((const float4*)in)[i];
    ushort4 o;
    o.x = f2bf(v.x); o.y = f2bf(v.y); o.z = f2bf(v.z); o.w = f2bf(v.w);
    ((ushort4*)out)[i] = o;
  }
}

// ---------------------------------------------------------------------------
// bf16 WMMA GEMM:  C[M,N] = A[M,K] * W[N,K]^T + bias[N]
// MODE 0: scatter bf16 into QKV head-split layout [3][B,H,T,D]
// MODE 1: plain fp32 row-major output
// Workgroup: 256 threads (8 waves), tile 128(M) x 64(N), K step 64.
// Tiles staged with async global->LDS DMA (ASYNCcnt) when available.
// ---------------------------------------------------------------------------
template <int MODE>
__global__ __launch_bounds__(256) void gemm_bf16_wmma(
    const unsigned short* __restrict__ A,
    const unsigned short* __restrict__ W,
    const float* __restrict__ bias,
    unsigned short* __restrict__ outBF,
    float* __restrict__ outF,
    int M, int N, int K) {
  __shared__ unsigned short As[128 * 72];   // row stride 72 elems (144B aligned)
  __shared__ unsigned short Bs[64 * 72];

  const int tid  = threadIdx.x;
  const int lane = tid & 31;
  const int wave = tid >> 5;
  const int gm0 = blockIdx.y * 128;
  const int gn0 = blockIdx.x * 64;

  v8f acc[4];
#pragma unroll
  for (int i = 0; i < 4; ++i) acc[i] = (v8f){0.f,0.f,0.f,0.f,0.f,0.f,0.f,0.f};

  // cooperative staging indices (K step 64)
  const int arowS = tid >> 1, akS = (tid & 1) * 32;   // A: 128 rows x 64 k
  const int browS = tid >> 2, bkS = (tid & 3) * 16;   // W: 64 rows x 64 k

  // fragment addressing (ISA 05_wmma layouts, wave32)
  const int fr_row = wave * 16 + (lane & 15);
  const int fr_akb = (lane < 16) ? 0 : 8;    // A: K base of first 8-chunk
  const int fr_bkb = (lane < 16) ? 0 : 8;    // B: K base of first 8-chunk

  for (int kk = 0; kk < K; kk += 64) {
    __syncthreads();
    {
      const unsigned short* src = A + (size_t)(gm0 + arowS) * K + kk + akS;
      stage16_async(src,      &As[arowS * 72 + akS]);
      stage16_async(src + 8,  &As[arowS * 72 + akS + 8]);
      stage16_async(src + 16, &As[arowS * 72 + akS + 16]);
      stage16_async(src + 24, &As[arowS * 72 + akS + 24]);
      const unsigned short* srcb = W + (size_t)(gn0 + browS) * K + kk + bkS;
      stage16_async(srcb,     &Bs[browS * 72 + bkS]);
      stage16_async(srcb + 8, &Bs[browS * 72 + bkS + 8]);
    }
    stage_wait();
    __syncthreads();

#pragma unroll
    for (int ks = 0; ks < 64; ks += 32) {
      FragBF a;
      a.q[0] = *(const uint4*)&As[fr_row * 72 + ks + fr_akb];
      a.q[1] = *(const uint4*)&As[fr_row * 72 + ks + fr_akb + 16];
#pragma unroll
      for (int nb = 0; nb < 4; ++nb) {
        FragBF b;
        int bcol = nb * 16 + (lane & 15);
        b.q[0] = *(const uint4*)&Bs[bcol * 72 + ks + fr_bkb];
        b.q[1] = *(const uint4*)&Bs[bcol * 72 + ks + fr_bkb + 16];
        acc[nb] = __builtin_amdgcn_wmma_f32_16x16x32_bf16(
            false, a.v, false, b.v, (short)0, acc[nb], false, false);
      }
    }
  }

  const int rowoff = (lane < 16) ? 0 : 8;
  const int coll   = lane & 15;
#pragma unroll
  for (int nb = 0; nb < 4; ++nb) {
    int gn = gn0 + nb * 16 + coll;
    float bv = bias[gn];
#pragma unroll
    for (int r = 0; r < 8; ++r) {
      int gm = gm0 + wave * 16 + rowoff + r;
      float v = acc[nb][r] + bv;
      if (MODE == 0) {
        int sel = gn >> 10, e = gn & 1023, h = e >> 6, dd = e & 63;
        int b = gm >> 11, t = gm & 2047;
        outBF[(size_t)sel * BHTD +
              (((size_t)(b * H_ + h) * T_ + t) * D_ + dd)] = f2bf(v);
      } else {
        outF[(size_t)gm * N + gn] = v;
      }
    }
  }
}

// ---------------------------------------------------------------------------
// Rotary: one thread per (b,h,t,pair). Rotates Q and K in place (bf16).
// Q is additionally pre-scaled by 1/sqrt(D) so the attention inner loop
// needs no per-tile score scaling.
// ---------------------------------------------------------------------------
__global__ __launch_bounds__(256) void rotary_kernel(
    unsigned short* __restrict__ qbuf, unsigned short* __restrict__ kbuf,
    const float* __restrict__ pos, const float* __restrict__ freqs) {
  int idx = blockIdx.x * blockDim.x + threadIdx.x;   // B*H*T*F = 4194304
  int f = idx & (F_ - 1);
  int t = (idx >> 5) & (T_ - 1);
  int h = (idx >> 16) & (H_ - 1);
  int b = idx >> 20;
  const float* p  = pos   + ((size_t)b * T_ + t) * 3;
  const float* fr = freqs + ((size_t)h * F_ + f) * 3;
  float ang = p[0] * fr[0] + p[1] * fr[1] + p[2] * fr[2];
  float s, c;
  __sincosf(ang, &s, &c);
  size_t base = ((size_t)(b * H_ + h) * T_ + t) * D_ + 2 * f;

  const float qscale = 0.125f;   // 1/sqrt(64)
  unsigned qv = *(const unsigned*)&qbuf[base];
  float q0 = bf2f((unsigned short)(qv & 0xFFFF)), q1 = bf2f((unsigned short)(qv >> 16));
  unsigned qo = (unsigned)f2bf((q0 * c - q1 * s) * qscale) |
                ((unsigned)f2bf((q0 * s + q1 * c) * qscale) << 16);
  *(unsigned*)&qbuf[base] = qo;

  unsigned kv = *(const unsigned*)&kbuf[base];
  float k0 = bf2f((unsigned short)(kv & 0xFFFF)), k1 = bf2f((unsigned short)(kv >> 16));
  unsigned ko = (unsigned)f2bf(k0 * c - k1 * s) | ((unsigned)f2bf(k0 * s + k1 * c) << 16);
  *(unsigned*)&kbuf[base] = ko;
}

// ---------------------------------------------------------------------------
// Flash attention, bf16 WMMA, online softmax with DPP row reductions.
// Grid: (T/128, B*H). Block: 256 threads = 8 waves; wave owns 16 query rows.
// Key tiles of 32; K tile staged via async global->LDS DMA, V tile goes
// through VGPRs (transposed on the way into LDS for the PV B-fragments).
// ---------------------------------------------------------------------------
__global__ __launch_bounds__(256) void flash_attn_wmma(
    const unsigned short* __restrict__ Q,
    const unsigned short* __restrict__ Kb,
    const unsigned short* __restrict__ V,
    unsigned short* __restrict__ O) {
  __shared__ unsigned short Ks[32 * 72];        // [key][d], stride 72 (144B aligned)
  __shared__ unsigned short Vt[64 * 40];        // transposed: [d][key], stride 40
  __shared__ unsigned short Pl[8 * 16 * 40];    // per-wave P tiles [16][32], stride 40

  const int tid  = threadIdx.x;
  const int lane = tid & 31;
  const int wave = tid >> 5;
  const int bh = blockIdx.y;
  const int b = bh >> 4, h = bh & 15;
  const int qt0 = blockIdx.x * 128;

  // Q fragments (A-matrix, 16x64 split into two 16x32); Q pre-scaled by 1/8.
  FragBF qf[2];
  {
    int row = qt0 + wave * 16 + (lane & 15);
    const unsigned short* qp = Q + ((size_t)bh * T_ + row) * D_;
#pragma unroll
    for (int dchunk = 0; dchunk < 2; ++dchunk) {
      int kb0 = dchunk * 32 + ((lane < 16) ? 0 : 8);
      qf[dchunk].q[0] = *(const uint4*)(qp + kb0);
      qf[dchunk].q[1] = *(const uint4*)(qp + kb0 + 16);
    }
  }

  float mrow[8], lrow[8];
  v8f acc[4];
#pragma unroll
  for (int r = 0; r < 8; ++r) { mrow[r] = -1e30f; lrow[r] = 0.f; }
#pragma unroll
  for (int i = 0; i < 4; ++i) acc[i] = (v8f){0.f,0.f,0.f,0.f,0.f,0.f,0.f,0.f};

  const int skey = tid >> 3;          // staging: 0..31 keys
  const int sd0  = (tid & 7) * 8;     // 8 contiguous d per thread
  const int rowoff = (lane < 16) ? 0 : 8;
  const int coll   = lane & 15;
  unsigned short* Pw = Pl + wave * 16 * 40;
  const v8f vzero = (v8f){0.f,0.f,0.f,0.f,0.f,0.f,0.f,0.f};

  for (int kt = 0; kt < T_; kt += 32) {
    __syncthreads();
    {
      const unsigned short* ksrc = Kb + ((size_t)bh * T_ + kt + skey) * D_ + sd0;
      stage16_async(ksrc, &Ks[skey * 72 + sd0]);        // K: async DMA to LDS
      const unsigned short* vsrc = V + ((size_t)bh * T_ + kt + skey) * D_ + sd0;
      uint4 vv = *(const uint4*)vsrc;                   // V: via VGPRs (transpose)
      unsigned short tmp[8];
      *(uint4*)tmp = vv;
#pragma unroll
      for (int j = 0; j < 8; ++j) Vt[(sd0 + j) * 40 + skey] = tmp[j];
    }
    stage_wait();
    __syncthreads();

    // S = Q * K^T  (16 x 32 tile = two 16x16 D-frags); scale pre-folded into Q
    v8f s0 = vzero, s1 = vzero;
#pragma unroll
    for (int dchunk = 0; dchunk < 2; ++dchunk) {
      FragBF kf0, kf1;
      int kdb = dchunk * 32 + ((lane < 16) ? 0 : 8);
      int kr0 = (lane & 15);
      int kr1 = 16 + (lane & 15);
      kf0.q[0] = *(const uint4*)&Ks[kr0 * 72 + kdb];
      kf0.q[1] = *(const uint4*)&Ks[kr0 * 72 + kdb + 16];
      kf1.q[0] = *(const uint4*)&Ks[kr1 * 72 + kdb];
      kf1.q[1] = *(const uint4*)&Ks[kr1 * 72 + kdb + 16];
      s0 = __builtin_amdgcn_wmma_f32_16x16x32_bf16(false, qf[dchunk].v, false, kf0.v,
                                                   (short)0, s0, false, false);
      s1 = __builtin_amdgcn_wmma_f32_16x16x32_bf16(false, qf[dchunk].v, false, kf1.v,
                                                   (short)0, s1, false, false);
    }

    // online softmax; 16-lane row reductions via DPP (no LDS traffic)
    float tmax[8], tsum[8], p0[8], p1[8], corr[8];
#pragma unroll
    for (int r = 0; r < 8; ++r) tmax[r] = fmaxf(s0[r], s1[r]);
    red16_max(tmax);
#pragma unroll
    for (int r = 0; r < 8; ++r) {
      float mnew = fmaxf(mrow[r], tmax[r]);
      corr[r] = __expf(mrow[r] - mnew);
      p0[r] = __expf(s0[r] - mnew);
      p1[r] = __expf(s1[r] - mnew);
      tsum[r] = p0[r] + p1[r];
      mrow[r] = mnew;
    }
    red16_sum(tsum);
#pragma unroll
    for (int r = 0; r < 8; ++r) lrow[r] = lrow[r] * corr[r] + tsum[r];
#pragma unroll
    for (int nb = 0; nb < 4; ++nb)
#pragma unroll
      for (int r = 0; r < 8; ++r) acc[nb][r] *= corr[r];

    // P (16x32 bf16) -> per-wave LDS tile, then reread as A-fragment
#pragma unroll
    for (int r = 0; r < 8; ++r) {
      Pw[(rowoff + r) * 40 + coll]      = f2bf(p0[r]);
      Pw[(rowoff + r) * 40 + 16 + coll] = f2bf(p1[r]);
    }
    FragBF pf;
    int prow = lane & 15;
    int pkb = (lane < 16) ? 0 : 8;
    pf.q[0] = *(const uint4*)&Pw[prow * 40 + pkb];
    pf.q[1] = *(const uint4*)&Pw[prow * 40 + pkb + 16];

    // O += P * V  (K-dim = 32 keys, N = 64 d -> 4 WMMAs)
#pragma unroll
    for (int nb = 0; nb < 4; ++nb) {
      FragBF vf;
      int vd  = nb * 16 + (lane & 15);
      int vkb = (lane < 16) ? 0 : 8;
      vf.q[0] = *(const uint4*)&Vt[vd * 40 + vkb];
      vf.q[1] = *(const uint4*)&Vt[vd * 40 + vkb + 16];
      acc[nb] = __builtin_amdgcn_wmma_f32_16x16x32_bf16(
          false, pf.v, false, vf.v, (short)0, acc[nb], false, false);
    }
  }

  // epilogue: normalize and write bf16 [B,T,E] (heads re-interleaved)
#pragma unroll
  for (int nb = 0; nb < 4; ++nb) {
#pragma unroll
    for (int r = 0; r < 8; ++r) {
      int tq = qt0 + wave * 16 + rowoff + r;
      float v = acc[nb][r] / lrow[r];
      O[((size_t)b * T_ + tq) * E_ + h * 64 + nb * 16 + coll] = f2bf(v);
    }
  }
}

// ---------------------------------------------------------------------------
extern "C" void kernel_launch(void* const* d_in, const int* in_sizes, int n_in,
                              void* d_out, int out_size, void* d_ws, size_t ws_size,
                              hipStream_t stream) {
  (void)in_sizes; (void)n_in; (void)out_size; (void)ws_size;
  const float* query     = (const float*)d_in[0];
  const float* positions = (const float*)d_in[1];
  const float* w_in      = (const float*)d_in[2];
  const float* b_in      = (const float*)d_in[3];
  const float* w_out     = (const float*)d_in[4];
  const float* b_out     = (const float*)d_in[5];
  const float* freqs     = (const float*)d_in[6];

  unsigned short* ws    = (unsigned short*)d_ws;
  unsigned short* qbf   = ws;                  // 8388608  (query bf16)
  unsigned short* wqkv  = qbf  + 8388608;      // 3145728  (in_proj bf16)
  unsigned short* woutb = wqkv + 3145728;      // 1048576  (out_proj bf16)
  unsigned short* qkv   = woutb + 1048576;     // 25165824 ([3][B,H,T,D] bf16)
  unsigned short* attn  = qkv  + 25165824;     // 8388608  ([B,T,E] bf16)

  f32_to_bf16_kernel<<<(8388608/4)/256, 256, 0, stream>>>(query, qbf, 8388608/4);
  f32_to_bf16_kernel<<<(3145728/4)/256, 256, 0, stream>>>(w_in, wqkv, 3145728/4);
  f32_to_bf16_kernel<<<(1048576/4)/256, 256, 0, stream>>>(w_out, woutb, 1048576/4);

  dim3 g1(3072 / 64, 8192 / 128);
  gemm_bf16_wmma<0><<<g1, 256, 0, stream>>>(qbf, wqkv, b_in, qkv, nullptr,
                                            8192, 3072, 1024);

  rotary_kernel<<<(B_*H_*T_*F_)/256, 256, 0, stream>>>(qkv, qkv + BHTD,
                                                       positions, freqs);

  dim3 g2(T_ / 128, B_ * H_);
  flash_attn_wmma<<<g2, 256, 0, stream>>>(qkv, qkv + BHTD, qkv + 2*(size_t)BHTD,
                                          attn);

  dim3 g3(1024 / 64, 8192 / 128);
  gemm_bf16_wmma<1><<<g3, 256, 0, stream>>>(attn, woutb, b_out, nullptr,
                                            (float*)d_out, 8192, 1024, 1024);
}
